// BiLSTM_CRF_35021163331585
// MI455X (gfx1250) — compile-verified, matrix operand order checked by reading, not run on hardware
//
#include <hip/hip_runtime.h>
#include <hip/hip_bf16.h>

// ---------------------------------------------------------------------------
// BiLSTM-CRF on gfx1250 (MI455X): bf16 WMMA GEMMs (async-LDS double buffered)
// + persistent LDS scan + CRF.   V=32000 T=256 B=32 E=512 H=256 L=2 K=17
// ---------------------------------------------------------------------------

typedef __bf16 bhalf;
typedef bhalf v16bf __attribute__((ext_vector_type(16)));
typedef bhalf v8bf  __attribute__((ext_vector_type(8)));
typedef float v8f   __attribute__((ext_vector_type(8)));

#define TSEQ 256
#define BSZ  32
#define EMB  512
#define HID  256
#define G4   1024     // 4*H
#define N2   2048     // both directions' gates stacked
#define MTOT 8192     // B*T
#define KTAG 17
#define HP   272      // padded LDS row stride for h (16B aligned, bank-spread)
#define PSA  48       // padded LDS row stride (bf16) for staged GEMM panels

#if __has_builtin(__builtin_amdgcn_global_load_async_to_lds_b128) && \
    __has_builtin(__builtin_amdgcn_s_wait_asynccnt)
#define USE_ASYNC_LDS 1
#else
#define USE_ASYNC_LDS 0
#endif

#if USE_ASYNC_LDS
// Builtin expects int4 pointers in AS(1) (prints as __device__) / AS(3).
typedef int v4i __attribute__((vector_size(16)));
typedef __attribute__((address_space(1))) v4i gv4i;
typedef __attribute__((address_space(3))) v4i lv4i;
__device__ __forceinline__ void async_cp16(const void* g, void* l) {
  // 16B global -> LDS DMA, tracked with ASYNCcnt
  __builtin_amdgcn_global_load_async_to_lds_b128((gv4i*)g, (lv4i*)l, 0, 0);
}
#endif

// Load one 16x16x32 bf16 WMMA fragment (A row-major or B=W[n][k] row-major).
// CDNA5 16-bit layout: lanes 0-15 hold K {0..7,16..23}, lanes 16-31 hold
// K {8..15,24..31} of the 32-wide K window; caller folds kSel into p.
__device__ __forceinline__ v16bf ld_frag(const bhalf* p) {
  v8bf lo = *reinterpret_cast<const v8bf*>(p);
  v8bf hi = *reinterpret_cast<const v8bf*>(p + 16);
  v16bf r;
#pragma unroll
  for (int i = 0; i < 8; ++i) { r[i] = lo[i]; r[i + 8] = hi[i]; }
  return r;
}

__device__ __forceinline__ v8f wmma_bf16(v16bf a, v16bf b, v8f c) {
  return __builtin_amdgcn_wmma_f32_16x16x32_bf16(false, a, false, b,
                                                 (short)0, c, false, false);
}

__device__ __forceinline__ float sigmoidf_fast(float x) {
  return 1.0f / (1.0f + __expf(-x));
}

// ---------------------------------------------------------------------------
// Kernel 1: embedding gather + f32 -> bf16 cast.  x0[m][e] = bf16(emb[src[m]][e])
// ---------------------------------------------------------------------------
__global__ void __launch_bounds__(256)
embed_cast(const int* __restrict__ src, const float* __restrict__ emb,
           bhalf* __restrict__ x0) {
  const size_t idx = (size_t)blockIdx.x * 256 + threadIdx.x;  // < 8192*512
  const int m = (int)(idx >> 9);
  const int e = (int)(idx & 511);
  const int tok = src[m];
  x0[idx] = (bhalf)emb[(size_t)tok * EMB + e];
}

// ---------------------------------------------------------------------------
// Kernel 2: cast all weights to bf16, combine biases.
// ---------------------------------------------------------------------------
__global__ void __launch_bounds__(256)
cast_weights(const float* __restrict__ Wih, const float* __restrict__ Whh,
             const float* __restrict__ bih, const float* __restrict__ bhh,
             bhalf* __restrict__ WihB, bhalf* __restrict__ WhhB,
             float* __restrict__ bias) {
  const size_t idx = (size_t)blockIdx.x * 256 + threadIdx.x;
  if (idx < (size_t)2 * 2 * G4 * EMB) WihB[idx] = (bhalf)Wih[idx];
  if (idx < (size_t)2 * 2 * G4 * HID) WhhB[idx] = (bhalf)Whh[idx];
  if (idx < (size_t)2 * 2 * G4) bias[idx] = bih[idx] + bhh[idx];
}

// ---------------------------------------------------------------------------
// Kernel 3: input projection GEMM.  XG[M=8192, N=2048] = X[M,512] @ W[2048,512]^T
// Block = 256 thr (8 waves), tile 128(M) x 64(N); wave tile 32x32 = 2x2 WMMA.
// Async path: double-buffered 128x32 A-panel + 64x32 B-panel staged to LDS
// with GLOBAL_LOAD_ASYNC_TO_LDS_B128; next panel DMA overlaps current WMMAs.
// ---------------------------------------------------------------------------
__global__ void __launch_bounds__(256)
gemm_xproj(const bhalf* __restrict__ X, const bhalf* __restrict__ W,
           float* __restrict__ XG) {
#if USE_ASYNC_LDS
  __shared__ bhalf As[2][128 * PSA];   // 2 x 12KB
  __shared__ bhalf Bs[2][64 * PSA];    // 2 x  6KB
#endif
  const int lane = threadIdx.x & 31;
  const int wid  = threadIdx.x >> 5;
  const int wm = wid & 3, wn = wid >> 2;
  const int mB = blockIdx.x * 128;
  const int nB = blockIdx.y * 64;
  const int r16 = lane & 15;
  const int kSel = (lane < 16) ? 0 : 8;
  const int rowOff = (lane >> 4) * 8;

  v8f acc[2][2];
#pragma unroll
  for (int mt = 0; mt < 2; ++mt)
#pragma unroll
    for (int nt = 0; nt < 2; ++nt)
#pragma unroll
      for (int e = 0; e < 8; ++e) acc[mt][nt][e] = 0.0f;

#if USE_ASYNC_LDS
  const int tid = threadIdx.x;
  // Stage one 32-wide K panel: A = 128 rows x 4 16B segs (2/thread),
  //                            B =  64 rows x 4 16B segs (1/thread).
#define STAGE_PANEL(bufi, kci)                                              \
  do {                                                                      \
    _Pragma("unroll")                                                       \
    for (int q = 0; q < 2; ++q) {                                           \
      const int s = q * 256 + tid;                                          \
      const int row = s >> 2, sg = s & 3;                                   \
      async_cp16(X + (size_t)(mB + row) * EMB + (kci) + sg * 8,             \
                 &As[(bufi)][row * PSA + sg * 8]);                          \
    }                                                                       \
    const int rowb = tid >> 2, sgb = tid & 3;                               \
    async_cp16(W + (size_t)(nB + rowb) * EMB + (kci) + sgb * 8,             \
               &Bs[(bufi)][rowb * PSA + sgb * 8]);                          \
  } while (0)

  STAGE_PANEL(0, 0);
  for (int kc = 0; kc < EMB; kc += 32) {
    const int buf = (kc >> 5) & 1;
    __builtin_amdgcn_s_wait_asynccnt(0);  // this wave's panel DMA done
    __syncthreads();                      // everyone's panel visible in LDS
    if (kc + 32 < EMB) STAGE_PANEL(buf ^ 1, kc + 32);
    const bhalf* Ab = &As[buf][0];
    const bhalf* Bb = &Bs[buf][0];
    v16bf a0 = ld_frag(Ab + (wm * 32 +      r16) * PSA + kSel);
    v16bf a1 = ld_frag(Ab + (wm * 32 + 16 + r16) * PSA + kSel);
    v16bf b0 = ld_frag(Bb + (wn * 32 +      r16) * PSA + kSel);
    v16bf b1 = ld_frag(Bb + (wn * 32 + 16 + r16) * PSA + kSel);
    acc[0][0] = wmma_bf16(a0, b0, acc[0][0]);
    acc[1][0] = wmma_bf16(a1, b0, acc[1][0]);
    acc[0][1] = wmma_bf16(a0, b1, acc[0][1]);
    acc[1][1] = wmma_bf16(a1, b1, acc[1][1]);
  }
#undef STAGE_PANEL
#else
  for (int kc = 0; kc < EMB; kc += 32) {
    const int kb = kc + kSel;
    v16bf a0 = ld_frag(X + (size_t)(mB + wm * 32 +      r16) * EMB + kb);
    v16bf a1 = ld_frag(X + (size_t)(mB + wm * 32 + 16 + r16) * EMB + kb);
    v16bf b0 = ld_frag(W + (size_t)(nB + wn * 32 +      r16) * EMB + kb);
    v16bf b1 = ld_frag(W + (size_t)(nB + wn * 32 + 16 + r16) * EMB + kb);
    acc[0][0] = wmma_bf16(a0, b0, acc[0][0]);
    acc[1][0] = wmma_bf16(a1, b0, acc[1][0]);
    acc[0][1] = wmma_bf16(a0, b1, acc[0][1]);
    acc[1][1] = wmma_bf16(a1, b1, acc[1][1]);
  }
#endif
  // C layout: VGPR r, lane l -> row = base + (l/16)*8 + r, col = base + l%16
#pragma unroll
  for (int mt = 0; mt < 2; ++mt)
#pragma unroll
    for (int nt = 0; nt < 2; ++nt) {
      const int col = nB + wn * 32 + nt * 16 + r16;
#pragma unroll
      for (int r = 0; r < 8; ++r) {
        const int row = mB + wm * 32 + mt * 16 + rowOff + r;
        XG[(size_t)row * N2 + col] = acc[mt][nt][r];
      }
    }
}

// ---------------------------------------------------------------------------
// Kernel 4: persistent LSTM scan.  grid = 2 (dir), block = 1024 thr (32 waves).
// Per step: g = h @ Whh^T via WMMA (each wave owns 32 of 1024 gate columns),
// result exchanged through LDS, fused gates, c in registers, h in LDS (bf16).
// Dynamic LDS: h[32][HP] bf16 + g[32][1024] f32  (~148.5 KB of 320 KB WGP LDS)
// ---------------------------------------------------------------------------
__global__ void __launch_bounds__(1024)
lstm_scan(const float* __restrict__ XG, const bhalf* __restrict__ WhhB,
          const float* __restrict__ bias, const int* __restrict__ src,
          bhalf* __restrict__ Xout) {
  extern __shared__ char smem[];
  bhalf* hS = reinterpret_cast<bhalf*>(smem);                       // 32 x HP
  float* gS = reinterpret_cast<float*>(smem + 32 * HP * sizeof(bhalf)); // 32x1024

  const int dir  = blockIdx.x;
  const int tid  = threadIdx.x;
  const int lane = tid & 31, wid = tid >> 5;
  const int r16 = lane & 15;
  const int kSel = (lane < 16) ? 0 : 8;
  const int rowOff = (lane >> 4) * 8;
  const int nBase = wid * 32;  // this wave's 32 gate columns within [0,1024)

  const float* XGd = XG   + (size_t)dir * G4;        // column offset in [M, 2048]
  const bhalf* Wd  = WhhB + (size_t)dir * G4 * HID;  // [1024, 256] bf16
  const float* bd  = bias + dir * G4;
  bhalf* Xod       = Xout + (size_t)dir * HID;       // this dir's half of concat

  // Warm Whh (L2-resident across the whole scan): ~512B per thread.
#pragma unroll
  for (int q = 0; q < 4; ++q)
    __builtin_prefetch(Wd + (size_t)tid * 256 + q * 64, 0, 1);

  for (int i = tid; i < 32 * HP; i += 1024) hS[i] = (bhalf)0.0f;
  float creg[8];
#pragma unroll
  for (int r = 0; r < 8; ++r) creg[r] = 0.0f;
  __syncthreads();

  for (int t = 0; t < TSEQ; ++t) {
    const int tt = dir ? (TSEQ - 1 - t) : t;

    // ---- phase 1: h @ Whh^T (M=32, N=32 per wave, K=256) ----
    v8f acc[2][2];
#pragma unroll
    for (int mt = 0; mt < 2; ++mt)
#pragma unroll
      for (int nt = 0; nt < 2; ++nt)
#pragma unroll
        for (int e = 0; e < 8; ++e) acc[mt][nt][e] = 0.0f;

#pragma unroll
    for (int ks = 0; ks < 8; ++ks) {
      const int kb = ks * 32 + kSel;
      v16bf a0 = ld_frag(hS + (     r16) * HP + kb);
      v16bf a1 = ld_frag(hS + (16 + r16) * HP + kb);
      v16bf b0 = ld_frag(Wd + (size_t)(nBase +      r16) * HID + kb);
      v16bf b1 = ld_frag(Wd + (size_t)(nBase + 16 + r16) * HID + kb);
      acc[0][0] = wmma_bf16(a0, b0, acc[0][0]);
      acc[1][0] = wmma_bf16(a1, b0, acc[1][0]);
      acc[0][1] = wmma_bf16(a0, b1, acc[0][1]);
      acc[1][1] = wmma_bf16(a1, b1, acc[1][1]);
    }
#pragma unroll
    for (int mt = 0; mt < 2; ++mt)
#pragma unroll
      for (int nt = 0; nt < 2; ++nt) {
        const int col = nBase + nt * 16 + r16;
#pragma unroll
        for (int r = 0; r < 8; ++r) {
          const int row = mt * 16 + rowOff + r;
          gS[row * G4 + col] = acc[mt][nt][r];
        }
      }
    __syncthreads();

    // ---- phase 2: fused gates; each thread owns 8 fixed (b, j) cells ----
#pragma unroll
    for (int rr = 0; rr < 8; ++rr) {
      const int idx = rr * 1024 + tid;        // 8192 = 32*256 cells
      const int b = idx >> 8, j = idx & 255;
      const float* xr = XGd + (size_t)(b * TSEQ + tt) * N2;
      const float gi = gS[b * G4 +       j] + xr[      j] + bd[      j];
      const float gf = gS[b * G4 + 256 + j] + xr[256 + j] + bd[256 + j];
      const float gc = gS[b * G4 + 512 + j] + xr[512 + j] + bd[512 + j];
      const float go = gS[b * G4 + 768 + j] + xr[768 + j] + bd[768 + j];
      const float iG = sigmoidf_fast(gi);
      const float fG = sigmoidf_fast(gf);
      const float gG = tanhf(gc);
      const float oG = sigmoidf_fast(go);
      const float cn = fG * creg[rr] + iG * gG;
      const float hn = oG * tanhf(cn);
      const bool msk = src[b * TSEQ + tt] != 0;
      const float hold = (float)hS[b * HP + j];
      creg[rr] = msk ? cn : creg[rr];
      Xod[(size_t)(b * TSEQ + tt) * (2 * HID) + j] = (bhalf)(msk ? hn : 0.0f);
      hS[b * HP + j] = (bhalf)(msk ? hn : hold);   // own-cell update only
    }
    __syncthreads();
  }
}

// ---------------------------------------------------------------------------
// Kernel 5: zero the scalar output (harness poisons d_out).
// ---------------------------------------------------------------------------
__global__ void zero_out(float* o) {
  if (threadIdx.x == 0 && blockIdx.x == 0) o[0] = 0.0f;
}

// ---------------------------------------------------------------------------
// Kernel 6: logits (K=17) + CRF forward NLL.  One block, 1024 thr:
// 32 groups of 32 lanes; lane j<17 owns tag j; alpha in LDS; logsumexp.
// ---------------------------------------------------------------------------
__global__ void __launch_bounds__(1024)
crf_nll(const bhalf* __restrict__ feats, const float* __restrict__ Wlin,
        const float* __restrict__ blin, const float* __restrict__ trans,
        const float* __restrict__ startv, const float* __restrict__ endv,
        const int* __restrict__ src, const int* __restrict__ labels,
        const int* __restrict__ lens, float* __restrict__ out) {
  __shared__ float Wl[KTAG * 512];
  __shared__ float alpha[BSZ][33];
  __shared__ float red[BSZ][33];
  __shared__ float trS[KTAG * KTAG];
  __shared__ float blS[KTAG], stS[KTAG], enS[KTAG];

  const int tid = threadIdx.x;
  for (int i = tid; i < KTAG * 512; i += 1024) Wl[i] = Wlin[i];
  for (int i = tid; i < KTAG * KTAG; i += 1024) trS[i] = trans[i];
  if (tid < KTAG) { blS[tid] = blin[tid]; stS[tid] = startv[tid]; enS[tid] = endv[tid]; }
  __syncthreads();

  const int b = tid >> 5;
  const int j = tid & 31;
  const bool act = (j < KTAG);
  const float NEG = -1e30f;
  float emit_acc = 0.0f, num0 = 0.0f;

  for (int t = 0; t < TSEQ; ++t) {
    float lg = NEG;
    if (act) {
      const bhalf* frow = feats + (size_t)(b * TSEQ + t) * 512;
      const float* wr = &Wl[j * 512];
      float s = blS[j];
#pragma unroll 8
      for (int k = 0; k < 512; ++k) s += (float)frow[k] * wr[k];
      lg = s;
    }
    const bool msk = src[b * TSEQ + t] != 0;
    const int lab = labels[b * TSEQ + t];
    if (act && msk && j == lab) emit_acc += lg;
    if (j == 0) {
      if (t == 0) num0 += stS[lab];
      else if (msk) num0 += trS[labels[b * TSEQ + t - 1] * KTAG + lab];
      if (t == lens[b] - 1) num0 += enS[lab];
    }
    if (t == 0) {
      if (act) alpha[b][j] = stS[j] + lg;
    } else {
      float newv = 0.0f;
      const bool doup = act && msk;
      if (doup) {
        float mx = NEG;
        for (int i2 = 0; i2 < KTAG; ++i2)
          mx = fmaxf(mx, alpha[b][i2] + trS[i2 * KTAG + j]);
        float sm = 0.0f;
        for (int i2 = 0; i2 < KTAG; ++i2)
          sm += __expf(alpha[b][i2] + trS[i2 * KTAG + j] - mx);
        newv = mx + __logf(sm) + lg;
      }
      __syncthreads();
      if (doup) alpha[b][j] = newv;
    }
    __syncthreads();
  }

  red[b][j] = act ? emit_acc : 0.0f;
  __syncthreads();
  if (j == 0) {
    float num = num0;
    for (int i2 = 0; i2 < KTAG; ++i2) num += red[b][i2];
    float mx = NEG;
    for (int i2 = 0; i2 < KTAG; ++i2) mx = fmaxf(mx, alpha[b][i2] + enS[i2]);
    float sm = 0.0f;
    for (int i2 = 0; i2 < KTAG; ++i2) sm += __expf(alpha[b][i2] + enS[i2] - mx);
    const float part = mx + __logf(sm);
    atomicAdd(out, part - num);   // loss = -(numerator - partition), summed
  }
}

// ---------------------------------------------------------------------------
// Host orchestration
// ---------------------------------------------------------------------------
extern "C" void kernel_launch(void* const* d_in, const int* in_sizes, int n_in,
                              void* d_out, int out_size, void* d_ws, size_t ws_size,
                              hipStream_t stream) {
  (void)in_sizes; (void)n_in; (void)out_size; (void)ws_size;
  const int*   src    = (const int*)d_in[0];
  const int*   lens   = (const int*)d_in[1];
  const int*   labels = (const int*)d_in[2];
  /* d_in[3] = decode (0) unused */
  const float* emb    = (const float*)d_in[4];
  const float* Wih    = (const float*)d_in[5];
  const float* Whh    = (const float*)d_in[6];
  const float* bih    = (const float*)d_in[7];
  const float* bhh    = (const float*)d_in[8];
  const float* Wlin   = (const float*)d_in[9];
  const float* blin   = (const float*)d_in[10];
  const float* trans  = (const float*)d_in[11];
  const float* startv = (const float*)d_in[12];
  const float* endv   = (const float*)d_in[13];
  float* out = (float*)d_out;

  char* ws = (char*)d_ws;
  size_t off = 0;
  bhalf* x0   = (bhalf*)(ws + off); off += (size_t)MTOT * EMB * sizeof(bhalf); // 8MB
  bhalf* x1   = (bhalf*)(ws + off); off += (size_t)MTOT * EMB * sizeof(bhalf); // 8MB
  bhalf* x2   = (bhalf*)(ws + off); off += (size_t)MTOT * EMB * sizeof(bhalf); // 8MB
  float* xg   = (float*)(ws + off); off += (size_t)MTOT * N2 * sizeof(float);  // 64MB
  bhalf* WihB = (bhalf*)(ws + off); off += (size_t)2 * 2 * G4 * EMB * sizeof(bhalf);
  bhalf* WhhB = (bhalf*)(ws + off); off += (size_t)2 * 2 * G4 * HID * sizeof(bhalf);
  float* bias = (float*)(ws + off); off += (size_t)2 * 2 * G4 * sizeof(float);

  // 1) embedding gather + bf16 cast
  embed_cast<<<(MTOT * EMB) / 256, 256, 0, stream>>>(src, emb, x0);
  // 2) weight casts + bias combine
  cast_weights<<<(2 * 2 * G4 * EMB) / 256, 256, 0, stream>>>(
      Wih, Whh, bih, bhh, WihB, WhhB, bias);

  const size_t scanSmem = (size_t)32 * HP * sizeof(bhalf) +
                          (size_t)32 * G4 * sizeof(float);

  // 3) layer 0: projection GEMM (both dirs) then persistent scan
  gemm_xproj<<<dim3(MTOT / 128, N2 / 64), 256, 0, stream>>>(x0, WihB, xg);
  lstm_scan<<<2, 1024, scanSmem, stream>>>(xg, WhhB, bias, src, x1);

  // 4) layer 1
  gemm_xproj<<<dim3(MTOT / 128, N2 / 64), 256, 0, stream>>>(
      x1, WihB + (size_t)2 * G4 * EMB, xg);
  lstm_scan<<<2, 1024, scanSmem, stream>>>(
      xg, WhhB + (size_t)2 * G4 * HID, bias + 2 * G4, src, x2);

  // 5) CRF NLL -> scalar
  zero_out<<<1, 1, 0, stream>>>(out);
  crf_nll<<<1, 1024, 0, stream>>>(x2, Wlin, blin, trans, startv, endv,
                                  src, labels, lens, out);
}
